// FlashInferLinear_6579889897967
// MI455X (gfx1250) — 8-bit matrix operand order repaired and checked by reading, compiled, not run on hardware
//
#include <hip/hip_runtime.h>

typedef __attribute__((ext_vector_type(16))) int   v16i;
typedef __attribute__((ext_vector_type(8)))  float v8f;

#define FP8_MAX 448.0f

__device__ __forceinline__ unsigned int f32_to_e4m3(float x) {
  union { float f; unsigned u; } v; v.f = x;
  unsigned sign = (v.u >> 24) & 0x80u;
  unsigned a    = v.u & 0x7FFFFFFFu;
  if (a > 0x7F800000u) return sign | 0x7Fu;
  int e = (int)(a >> 23) - 127;
  if (e < -9) return sign;
  if (e < -6) {
    float s = __uint_as_float(a) * 512.0f;
    int m = (int)rintf(s);
    if (m >= 8) return sign | 0x08u;
    return sign | (unsigned)m;
  }
  unsigned mant = a & 0x7FFFFFu;
  unsigned keep = mant >> 20;
  unsigned rest = mant & 0xFFFFFu;
  keep += (rest > 0x80000u) || ((rest == 0x80000u) && (keep & 1u));
  unsigned ebits = (unsigned)(e + 7);
  if (keep == 8u) { keep = 0u; ebits += 1u; }
  unsigned code = (ebits << 3) | keep;
  if (code > 0x7Eu) code = 0x7Eu;
  return sign | code;
}

__global__ void fi_init_kernel(unsigned* amax) {
  amax[0] = 0u; amax[1] = 0u;
}

__global__ __launch_bounds__(256)
void fi_amax_kernel(const float* __restrict__ x, size_t n, unsigned* __restrict__ amax) {
  __shared__ unsigned red[256];
  float m = 0.0f;
  size_t stride = (size_t)gridDim.x * blockDim.x;
  for (size_t i = (size_t)blockIdx.x * blockDim.x + threadIdx.x; i < n; i += stride)
    m = fmaxf(m, fabsf(x[i]));
  red[threadIdx.x] = __float_as_uint(m);
  __syncthreads();
  for (int s = 128; s > 0; s >>= 1) {
    if ((int)threadIdx.x < s)
      red[threadIdx.x] = max(red[threadIdx.x], red[threadIdx.x + s]);
    __syncthreads();
  }
  if (threadIdx.x == 0) atomicMax(amax, red[0]);
}

__global__ void fi_scale_kernel(const unsigned* __restrict__ amax, float* __restrict__ scales) {
  float ax = fmaxf(__uint_as_float(amax[0]), 1e-12f);
  float aw = fmaxf(__uint_as_float(amax[1]), 1e-12f);
  scales[0] = ax / FP8_MAX;
  scales[1] = aw / FP8_MAX;
  scales[2] = FP8_MAX / ax;
  scales[3] = FP8_MAX / aw;
}

__global__ __launch_bounds__(256)
void fi_quant_kernel(const float* __restrict__ x, unsigned char* __restrict__ q,
                     size_t n4, const float* __restrict__ invscale) {
  const float is = *invscale;
  size_t stride = (size_t)gridDim.x * blockDim.x;
  for (size_t i = (size_t)blockIdx.x * blockDim.x + threadIdx.x; i < n4; i += stride) {
    const float4 v = ((const float4*)x)[i];
    unsigned b0 = f32_to_e4m3(v.x * is);
    unsigned b1 = f32_to_e4m3(v.y * is);
    unsigned b2 = f32_to_e4m3(v.z * is);
    unsigned b3 = f32_to_e4m3(v.w * is);
    ((unsigned int*)q)[i] = b0 | (b1 << 8) | (b2 << 16) | (b3 << 24);
  }
}

__global__ __launch_bounds__(256)
void fi_fp8_gemm_wmma(const unsigned char* __restrict__ Aq,
                      const unsigned char* __restrict__ Wq,
                      const float* __restrict__ bias,
                      const float* __restrict__ scales,
                      float* __restrict__ C,
                      int M, int N, int K) {
  __shared__ unsigned char smem[2][32768];

  const int lane  = threadIdx.x & 31;
  const int wave  = threadIdx.x >> 5;
  const int row16 = lane & 15;
  const int hi    = lane >> 4;

  const int bm = blockIdx.x * 128;
  const int bn = blockIdx.y * 128;
  const int wm = (wave & 1) * 64;
  const int wn = (wave >> 1) * 32;

  const unsigned char* gA = Aq + (size_t)(bm + wave * 16 + row16) * K + hi * 16;
  const unsigned char* gB = Wq + (size_t)(bn + wave * 16 + row16) * K + hi * 16;

  const unsigned smemBase = (unsigned)(uintptr_t)&smem[0][0];
  const unsigned aDst0 = smemBase + (unsigned)(wave * 2048 + lane * 16);
  const unsigned bDst0 = smemBase + 16384u + (unsigned)(wave * 2048 + lane * 16);

  auto stage = [&](int buf, int k0) {
    const unsigned bufOff = (unsigned)buf * 32768u;
    const unsigned char* ga = gA + k0;
#pragma unroll
    for (int c = 0; c < 8; ++c) {
      asm volatile("global_load_async_to_lds_b64 %0, %1, off"
                   :: "v"(aDst0 + bufOff + (unsigned)((c >> 1) * 512 + (c & 1) * 8)),
                      "v"(ga + (c >> 1) * 32 + (c & 1) * 8)
                   : "memory");
    }
    const unsigned char* gb = gB + k0;
#pragma unroll
    for (int q = 0; q < 4; ++q) {
      asm volatile("global_load_async_to_lds_b128 %0, %1, off"
                   :: "v"(bDst0 + bufOff + (unsigned)(q * 512)),
                      "v"(gb + q * 32)
                   : "memory");
    }
  };

  v8f acc[4][2] = {};

  const int kIters = K >> 7;
  stage(0, 0);
  for (int it = 0; it < kIters; ++it) {
    const int cur = it & 1;
    if (it + 1 < kIters) {
      stage(cur ^ 1, (it + 1) << 7);
      asm volatile("s_wait_asynccnt 12" ::: "memory");
    } else {
      asm volatile("s_wait_asynccnt 0" ::: "memory");
    }
    __syncthreads();

    const unsigned char* bufA = &smem[cur][0];
    const unsigned char* bufB = &smem[cur][16384];

    v16i afrag[4];
#pragma unroll
    for (int t = 0; t < 4; ++t) {
      const unsigned char* pa = bufA + ((wm >> 4) + t) * 2048 + lane * 16;
#pragma unroll
      for (int i = 0; i < 4; ++i) {
        const uint4 d = *(const uint4*)(pa + i * 512);
        afrag[t][4 * i + 0] = (int)d.x;
        afrag[t][4 * i + 1] = (int)d.y;
        afrag[t][4 * i + 2] = (int)d.z;
        afrag[t][4 * i + 3] = (int)d.w;
      }
    }
    v16i bfrag[2];
#pragma unroll
    for (int u = 0; u < 2; ++u) {
      const unsigned char* pb = bufB + ((wn >> 4) + u) * 2048 + lane * 16;
#pragma unroll
      for (int q = 0; q < 4; ++q) {
        const uint4 d = *(const uint4*)(pb + q * 512);
        bfrag[u][4 * q + 0] = (int)d.x;
        bfrag[u][4 * q + 1] = (int)d.y;
        bfrag[u][4 * q + 2] = (int)d.z;
        bfrag[u][4 * q + 3] = (int)d.w;
      }
    }

#pragma unroll
    for (int t = 0; t < 4; ++t) {
#pragma unroll
      for (int u = 0; u < 2; ++u) {
        acc[t][u] = __builtin_amdgcn_wmma_f32_16x16x128_fp8_fp8(
            afrag[t], bfrag[u], (short)0, acc[t][u], false, false);
      }
    }
    __syncthreads();
  }

  const float s = scales[0] * scales[1];
#pragma unroll
  for (int u = 0; u < 2; ++u) {
    const int n = bn + wn + u * 16 + row16;
    const float bv = bias[n];
#pragma unroll
    for (int t = 0; t < 4; ++t) {
      const int mbase = bm + wm + t * 16 + hi * 8;
#pragma unroll
      for (int r = 0; r < 8; ++r) {
        C[(size_t)(mbase + r) * N + n] = acc[t][u][r] * s + bv;
      }
    }
  }
}

extern "C" void kernel_launch(void* const* d_in, const int* in_sizes, int n_in,
                              void* d_out, int out_size, void* d_ws, size_t ws_size,
                              hipStream_t stream) {
  const float* x    = (const float*)d_in[0];
  const float* w    = (const float*)d_in[1];
  const float* bias = (const float*)d_in[2];
  float* out = (float*)d_out;

  const int N = in_sizes[2];
  const int K = in_sizes[1] / N;
  const int M = in_sizes[0] / K;
  const size_t nx = (size_t)M * K;
  const size_t nw = (size_t)N * K;

  unsigned char* ws = (unsigned char*)d_ws;
  unsigned* amax  = (unsigned*)ws;
  float*   scales = (float*)(ws + 16);
  unsigned char* Aq = ws + 256;
  unsigned char* Wq = Aq + nx;

  fi_init_kernel<<<1, 1, 0, stream>>>(amax);

  fi_amax_kernel<<<2048, 256, 0, stream>>>(x, nx, &amax[0]);
  fi_amax_kernel<<<2048, 256, 0, stream>>>(w, nw, &amax[1]);
  fi_scale_kernel<<<1, 1, 0, stream>>>(amax, scales);

  fi_quant_kernel<<<4096, 256, 0, stream>>>(x, Aq, nx / 4, &scales[2]);
  fi_quant_kernel<<<4096, 256, 0, stream>>>(w, Wq, nw / 4, &scales[3]);

  dim3 grid(M / 128, N / 128);
  fi_fp8_gemm_wmma<<<grid, 256, 0, stream>>>(Aq, Wq, bias, scales, out, M, N, K);
}
